// Q_AlexOutputBlock_49529562857866
// MI455X (gfx1250) — compile-verified, matrix operand order checked by reading, not run on hardware
//
#include <hip/hip_runtime.h>

typedef __attribute__((ext_vector_type(8))) int v8i;

// ---------------------------------------------------------------------------
// Scale utilities
// ---------------------------------------------------------------------------
__global__ void k_zero(float* p, int n) {
  int i = blockIdx.x * blockDim.x + threadIdx.x;
  if (i < n) p[i] = 0.0f;
}

// max(|x|) over nv4 float4's -> atomic max into *slot (float bits as u32; abs>=0)
__global__ __launch_bounds__(256) void k_maxabs(const float* __restrict__ x,
                                                float* slot, long nv4) {
  const float4* x4 = (const float4*)x;
  float m = 0.f;
  long stride = (long)gridDim.x * blockDim.x;
  for (long i = blockIdx.x * (long)blockDim.x + threadIdx.x; i < nv4; i += stride) {
    float4 v = x4[i];
    m = fmaxf(m, fmaxf(fmaxf(fabsf(v.x), fabsf(v.y)),
                       fmaxf(fabsf(v.z), fabsf(v.w))));
  }
  __shared__ float red[256];
  red[threadIdx.x] = m;
  __syncthreads();
  for (int s = 128; s > 0; s >>= 1) {
    if ((int)threadIdx.x < s)
      red[threadIdx.x] = fmaxf(red[threadIdx.x], red[threadIdx.x + s]);
    __syncthreads();
  }
  if (threadIdx.x == 0)
    atomicMax((unsigned int*)slot, __float_as_uint(red[0]));
}

// q = clamp(rint(x * 127/max), lo, 127) packed 4x int8 per dword
__global__ __launch_bounds__(256) void k_quant(const float* __restrict__ x,
                                               char* __restrict__ q,
                                               const float* __restrict__ slot,
                                               float lo, long nv4) {
  float mx = *slot;
  float inv = mx > 0.f ? 127.0f / mx : 0.f;
  const float4* x4 = (const float4*)x;
  unsigned int* q4 = (unsigned int*)q;
  long stride = (long)gridDim.x * blockDim.x;
  for (long i = blockIdx.x * (long)blockDim.x + threadIdx.x; i < nv4; i += stride) {
    float4 v = x4[i];
    int q0 = (int)fminf(fmaxf(rintf(v.x * inv), lo), 127.f);
    int q1 = (int)fminf(fmaxf(rintf(v.y * inv), lo), 127.f);
    int q2 = (int)fminf(fmaxf(rintf(v.z * inv), lo), 127.f);
    int q3 = (int)fminf(fmaxf(rintf(v.w * inv), lo), 127.f);
    q4[i] = (unsigned)(q0 & 0xff) | ((unsigned)(q1 & 0xff) << 8) |
            ((unsigned)(q2 & 0xff) << 16) | ((unsigned)(q3 & 0xff) << 24);
  }
}

// ---------------------------------------------------------------------------
// Int8 GEMM with V_WMMA_I32_16X16X64_IU8 + async Global->LDS double buffering.
//   A: [M x K] int8 row-major (activations), W: [N x K] int8 row-major.
//   out[m,n] = ((i32)sum_k A[m,k]*W[n,k] + rint(b[n]/sb)) * sb, sb=(mA/127)(mW/127)
// Block: 128 threads (4 wave32), tile 64(M) x 128(N), K stepped by 64.
// LDS row stride = 20 dwords (80B): 16B-aligned async stores, and 20*r mod 64
// distinct for r=0..15 -> conflict-free fragment reads.
// Global->LDS uses GLOBAL_LOAD_ASYNC_TO_LDS_B128 (ASYNCcnt) so tile data never
// transits VGPRs; wave waits s_wait_asynccnt then workgroup barrier.
// ---------------------------------------------------------------------------
#define LDSROW 20
#define LDSROWB 80

template <bool RELU>
__global__ __launch_bounds__(128) void k_gemm_i8(
    const char* __restrict__ Aq, const char* __restrict__ Wq,
    const float* __restrict__ bias, const float* __restrict__ maxA,
    const float* __restrict__ maxW, float* __restrict__ out,
    float* maxOut, int M, int N, int K) {
  __shared__ int lA[2][64 * LDSROW];
  __shared__ int lB[2][128 * LDSROW];

  const int tid = threadIdx.x;
  const int lane = tid & 31;
  const int w = tid >> 5;      // wave 0..3 -> M strip
  const int g = lane >> 4;     // lane half of the 8-bit K swizzle
  const int r = lane & 15;

  const int m0 = blockIdx.y * 64;
  const int n0 = blockIdx.x * 128;
  const int nk = K >> 6;

  v8i acc[8] = {};

  // LDS byte offsets (generic->LDS: low 32 bits are the wave-relative offset)
  const unsigned ldsA0 = (unsigned)(size_t)(void*)&lA[0][0];
  const unsigned ldsA1 = (unsigned)(size_t)(void*)&lA[1][0];
  const unsigned ldsB0 = (unsigned)(size_t)(void*)&lB[0][0];
  const unsigned ldsB1 = (unsigned)(size_t)(void*)&lB[1][0];

  // Issue this thread's 6 async 16B chunks (2 for A-tile, 4 for B-tile).
  auto issueLoads = [&](int buf, int kb) {
    const unsigned ldsA = buf ? ldsA1 : ldsA0;
    const unsigned ldsB = buf ? ldsB1 : ldsB0;
#pragma unroll
    for (int i = 0; i < 2; ++i) {
      int c = tid + i * 128, row = c >> 2, cc = c & 3;
      unsigned dst = ldsA + (unsigned)(row * LDSROWB + cc * 16);
      unsigned long long src =
          (unsigned long long)(size_t)(Aq + (size_t)(m0 + row) * K +
                                       (size_t)kb * 64 + cc * 16);
      asm volatile("global_load_async_to_lds_b128 %0, %1, off"
                   :: "v"(dst), "v"(src));
    }
#pragma unroll
    for (int i = 0; i < 4; ++i) {
      int c = tid + i * 128, row = c >> 2, cc = c & 3;
      int n = n0 + row;
      if (n >= N) n = N - 1;  // clamp: stays in-bounds, column never stored
      unsigned dst = ldsB + (unsigned)(row * LDSROWB + cc * 16);
      unsigned long long src =
          (unsigned long long)(size_t)(Wq + (size_t)n * K +
                                       (size_t)kb * 64 + cc * 16);
      asm volatile("global_load_async_to_lds_b128 %0, %1, off"
                   :: "v"(dst), "v"(src));
    }
  };

  issueLoads(0, 0);

  for (int kb = 0; kb < nk; ++kb) {
    const int cur = kb & 1;
    if (kb + 1 < nk) {
      issueLoads(cur ^ 1, kb + 1);
      // 6 chunks for the next buffer may stay in flight; current buffer done.
      asm volatile("s_wait_asynccnt 0x6" ::: "memory");
    } else {
      asm volatile("s_wait_asynccnt 0x0" ::: "memory");
    }
    __syncthreads();  // all waves' async fills for `cur` have landed

    // ISA 8-bit fragment layout: dword v, lane-half g -> K dword offset
    //   v0:2g v1:2g+1 v2:2g+4 v3:2g+5 v4:2g+8 v5:2g+9 v6:2g+12 v7:2g+13
    const int* pA = &lA[cur][(w * 16 + r) * LDSROW + g * 2];
    v8i a;
    a[0] = pA[0];  a[1] = pA[1];  a[2] = pA[4];  a[3] = pA[5];
    a[4] = pA[8];  a[5] = pA[9];  a[6] = pA[12]; a[7] = pA[13];
#pragma unroll
    for (int j = 0; j < 8; ++j) {
      const int* pB = &lB[cur][(j * 16 + r) * LDSROW + g * 2];
      v8i b;
      b[0] = pB[0];  b[1] = pB[1];  b[2] = pB[4];  b[3] = pB[5];
      b[4] = pB[8];  b[5] = pB[9];  b[6] = pB[12]; b[7] = pB[13];
      acc[j] = __builtin_amdgcn_wmma_i32_16x16x64_iu8(
          /*sgn_a=*/true, a, /*sgn_b=*/true, b, acc[j],
          /*reuse_a=*/false, /*reuse_b=*/false);
    }
    __syncthreads();  // all waves done reading `cur` before it is refilled
  }

  // Epilogue: dequant + quantized bias + optional ReLU + next-layer max
  float sA = maxA[0] * (1.0f / 127.0f);
  float sW = maxW[0] * (1.0f / 127.0f);
  float sb = sA * sW;
  float inv_sb = sb > 0.f ? 1.0f / sb : 0.f;

  float mloc = 0.f;
  const int mbase = m0 + w * 16 + g * 8;  // C layout: VGPR i -> M = i + 8g
#pragma unroll
  for (int j = 0; j < 8; ++j) {
    int n = n0 + j * 16 + r;  // C layout: N = lane%16
    if (n >= N) continue;
    float bq = rintf(bias[n] * inv_sb);
#pragma unroll
    for (int i = 0; i < 8; ++i) {
      float v = ((float)acc[j][i] + bq) * sb;
      if (RELU) v = fmaxf(v, 0.f);
      out[(size_t)(mbase + i) * N + n] = v;
      mloc = fmaxf(mloc, v);
    }
  }
  if (RELU && maxOut)  // relu output >= 0 => float bits monotonic as u32
    atomicMax((unsigned int*)maxOut, __float_as_uint(mloc));
}

// ---------------------------------------------------------------------------
extern "C" void kernel_launch(void* const* d_in, const int* in_sizes, int n_in,
                              void* d_out, int out_size, void* d_ws, size_t ws_size,
                              hipStream_t stream) {
  const float* x  = (const float*)d_in[0];
  const float* w1 = (const float*)d_in[1];
  const float* b1 = (const float*)d_in[2];
  const float* w2 = (const float*)d_in[3];
  const float* b2 = (const float*)d_in[4];
  const float* w3 = (const float*)d_in[5];
  const float* b3 = (const float*)d_in[6];
  float* out = (float*)d_out;

  const int B = 512, DIN = 9216, DH = 4096, DOUT = 1000;

  char* ws = (char*)d_ws;
  size_t off = 0;
  auto alloc = [&](size_t bytes) {
    size_t o = off;
    off += (bytes + 255) & ~(size_t)255;
    return o;
  };
  float* scales = (float*)(ws + alloc(256));  // [0]=mx x,[1..3]=mx w1..w3,[4]=mx h1,[5]=mx h2
  char* Xq  = ws + alloc((size_t)B * DIN);
  char* W1q = ws + alloc((size_t)DH * DIN);
  char* W2q = ws + alloc((size_t)DH * DH);
  char* W3q = ws + alloc((size_t)DOUT * DH);
  float* H1 = (float*)(ws + alloc((size_t)B * DH * 4));
  float* H2 = (float*)(ws + alloc((size_t)B * DH * 4));
  char* H1q = ws + alloc((size_t)B * DH);
  char* H2q = ws + alloc((size_t)B * DH);

  k_zero<<<1, 32, 0, stream>>>(scales, 8);

  auto rgrid = [](long nv4) {
    long b = (nv4 + 255) / 256;
    return (int)(b > 4096 ? 4096 : b);
  };
  const long nx4 = (long)B * DIN / 4,  nw14 = (long)DH * DIN / 4;
  const long nw24 = (long)DH * DH / 4, nw34 = (long)DOUT * DH / 4;
  const long nh4 = (long)B * DH / 4;

  k_maxabs<<<rgrid(nx4),  256, 0, stream>>>(x,  scales + 0, nx4);
  k_maxabs<<<rgrid(nw14), 256, 0, stream>>>(w1, scales + 1, nw14);
  k_maxabs<<<rgrid(nw24), 256, 0, stream>>>(w2, scales + 2, nw24);
  k_maxabs<<<rgrid(nw34), 256, 0, stream>>>(w3, scales + 3, nw34);

  k_quant<<<rgrid(nx4),  256, 0, stream>>>(x,  Xq,  scales + 0, -128.f, nx4);
  k_quant<<<rgrid(nw14), 256, 0, stream>>>(w1, W1q, scales + 1, -127.f, nw14);
  k_quant<<<rgrid(nw24), 256, 0, stream>>>(w2, W2q, scales + 2, -127.f, nw24);
  k_quant<<<rgrid(nw34), 256, 0, stream>>>(w3, W3q, scales + 3, -127.f, nw34);

  dim3 blk(128);
  k_gemm_i8<true><<<dim3(DH / 128, B / 64), blk, 0, stream>>>(
      Xq, W1q, b1, scales + 0, scales + 1, H1, scales + 4, B, DH, DIN);
  k_quant<<<rgrid(nh4), 256, 0, stream>>>(H1, H1q, scales + 4, -128.f, nh4);

  k_gemm_i8<true><<<dim3(DH / 128, B / 64), blk, 0, stream>>>(
      H1q, W2q, b2, scales + 4, scales + 2, H2, scales + 5, B, DH, DH);
  k_quant<<<rgrid(nh4), 256, 0, stream>>>(H2, H2q, scales + 5, -128.f, nh4);

  k_gemm_i8<false><<<dim3((DOUT + 127) / 128, B / 64), blk, 0, stream>>>(
      H2q, W3q, b3, scales + 5, scales + 3, out, nullptr, B, DOUT, DH);
}
